// MoE_53987738911302
// MI455X (gfx1250) — matrix operand order repaired and checked by reading, not compiled, not run on hardware
//
#include <hip/hip_runtime.h>
#include <hip/hip_bf16.h>
#include <math.h>

// ---------------------------------------------------------------------------
// MoE forward for MI455X (gfx1250, wave32, WMMA + async-to-LDS).
// Big GEMMs: bf16 WMMA (v_wmma_f32_16x16x32_bf16), fp32 accumulate,
// double-buffered LDS, B tiles streamed with GLOBAL_LOAD_ASYNC_TO_LDS_B128,
// A tiles converted fp32->bf16 with native packed casts while staging.
// ---------------------------------------------------------------------------

#define BT   4096   // batch
#define OBSD 1024
#define HIDD 1024
#define LATD 512
#define OUTD 19
#define NEXP 8
#define NBLK 3
#define EPSF 2.220446049250313e-16f

#define GEMM_USE_ASYNC 1   // gfx1250 async-to-LDS path (ASYNCcnt); 0 = plain loads

typedef __bf16 bf16x16 __attribute__((ext_vector_type(16)));
typedef __bf16 bf16x4  __attribute__((ext_vector_type(4)));
typedef float  f32x8   __attribute__((ext_vector_type(8)));

union Frag { unsigned int d[8]; bf16x16 v; };

#if GEMM_USE_ASYNC
// LDS byte-offset of a generic pointer to a __shared__ object (as(3) ptrtoint).
__device__ __forceinline__ unsigned lds_off(const void* p) {
  return (unsigned)(unsigned long long)(__attribute__((address_space(3))) const void*)p;
}
// GLOBAL_LOAD_ASYNC_TO_LDS_B128, GV mode (64-bit vaddr, saddr=off). ASYNCcnt.
__device__ __forceinline__ void async_g2l_b128(unsigned loff, const void* gaddr) {
  asm volatile("global_load_async_to_lds_b128 %0, %1, off"
               :: "v"(loff), "v"(gaddr) : "memory");
}
__device__ __forceinline__ void wait_async0() {
  asm volatile("s_wait_asynccnt 0x0" ::: "memory");
}
#endif

// ---------------------------------------------------------------------------
// Transpose + convert: W[L][Kd][N] fp32 -> Wt[L][N][Kd] bf16 (32x32 LDS tiles)
// ---------------------------------------------------------------------------
__global__ __launch_bounds__(256)
void transpose_bf16(const float* __restrict__ W, __bf16* __restrict__ Wt,
                    int Kd, int N)
{
  __shared__ float tile[32][33];
  const size_t mat = (size_t)blockIdx.z * Kd * N;
  const int k0 = blockIdx.y * 32, n0 = blockIdx.x * 32;
  const int tx = threadIdx.x & 31, ty = threadIdx.x >> 5;   // 32x8
  #pragma unroll
  for (int i = 0; i < 32; i += 8)
    tile[ty + i][tx] = W[mat + (size_t)(k0 + ty + i) * N + n0 + tx];
  __syncthreads();
  #pragma unroll
  for (int i = 0; i < 32; i += 8)
    Wt[mat + (size_t)(n0 + ty + i) * Kd + k0 + tx] = (__bf16)tile[tx][ty + i];
}

// ---------------------------------------------------------------------------
// C[M,N] = act(A[M,K] @ W[K,N] + bias[N]) (+ resid[M,N])
//   A:  fp32 [M,K]   (converted to bf16 while staging to LDS)
//   Wt: bf16 [N,K]   (pre-transposed; B tile is a raw byte copy -> async DMA)
// Block tile 128x128, 8 waves (4x2), wave = 2x4 grid of 16x16x32 WMMA tiles.
// Double-buffered LDS; prefetch tile t+1 while computing tile t.
// M%128==0, N%128==0, K%32==0 for all launches here.
// ---------------------------------------------------------------------------
__global__ __launch_bounds__(256)
void gemm_bias_act(const float* __restrict__ A, const __bf16* __restrict__ Wt,
                   const float* __restrict__ bias, const float* __restrict__ resid,
                   float* __restrict__ C, int M, int N, int Kd, int relu)
{
  __shared__ __bf16 sA[2][128][36];  // [buf][row][k]  bf16, padded
  __shared__ __bf16 sB[2][128][36];  // [buf][col][k]  bf16, padded

  const int tid  = threadIdx.x;
  const int lane = tid & 31;
  const int wave = tid >> 5;
  const int wm   = wave >> 1;     // 0..3  -> 32-row strip
  const int wn   = wave & 1;      // 0..1  -> 64-col strip
  const int hfl  = lane >> 4;     // half-wave (K-subrange select)
  const int lr   = lane & 15;     // row (A) / col (B) within 16x16 tile
  const int bm   = blockIdx.y * 128;
  const int bn   = blockIdx.x * 128;

  f32x8 acc[2][4];
  #pragma unroll
  for (int mi = 0; mi < 2; ++mi)
    #pragma unroll
    for (int ni = 0; ni < 4; ++ni)
      #pragma unroll
      for (int j = 0; j < 8; ++j) acc[mi][ni][j] = 0.0f;

  float4 fa[4];                  // A-tile prefetch registers
#if !GEMM_USE_ASYNC
  uint4  fb[2];                  // B-tile prefetch registers (fallback path)
#endif

  // ---- B tile: 128 cols x 32 k of bf16 = 8KB; 2 x b128 chunks per thread ----
  auto issueB = [&](int buf, int kt) {
    #pragma unroll
    for (int i = 0; i < 2; ++i) {
      const int c   = tid + i * 256;       // chunk id 0..511
      const int row = c >> 2;              // col 0..127
      const int k8  = (c & 3) << 3;        // k offset {0,8,16,24}
      const __bf16* g = Wt + (size_t)(bn + row) * Kd + kt + k8;
#if GEMM_USE_ASYNC
      async_g2l_b128(lds_off(&sB[buf][row][k8]), g);
#else
      fb[i] = *(const uint4*)g;
#endif
    }
  };
#if !GEMM_USE_ASYNC
  auto commitB = [&](int buf) {
    #pragma unroll
    for (int i = 0; i < 2; ++i) {
      const int c   = tid + i * 256;
      const int row = c >> 2;
      const int k8  = (c & 3) << 3;
      *(uint4*)&sB[buf][row][k8] = fb[i];
    }
  };
#endif

  // ---- A tile: 128 rows x 32 k fp32; 4 float4 per thread, convert on store ----
  auto loadA = [&](int kt) {
    #pragma unroll
    for (int i = 0; i < 4; ++i) {
      const int s = tid + i * 256, r = s >> 3, c4 = (s & 7) << 2;
      fa[i] = *(const float4*)(A + (size_t)(bm + r) * Kd + kt + c4);
    }
  };
  auto storeA = [&](int buf) {
    #pragma unroll
    for (int i = 0; i < 4; ++i) {
      const int s = tid + i * 256, r = s >> 3, c4 = (s & 7) << 2;
      bf16x4 q;                            // native fptrunc -> v_cvt_pk_bf16_f32
      q[0] = (__bf16)fa[i].x;
      q[1] = (__bf16)fa[i].y;
      q[2] = (__bf16)fa[i].z;
      q[3] = (__bf16)fa[i].w;
      *(bf16x4*)&sA[buf][r][c4] = q;       // 8B-aligned ds_store_b64
    }
  };

  // ---- prologue: stage tile 0 into buffer 0 ----
  issueB(0, 0);
  loadA(0);
#if !GEMM_USE_ASYNC
  commitB(0);
#endif
  storeA(0);
#if GEMM_USE_ASYNC
  wait_async0();
#endif
  __syncthreads();

  const int ntiles = Kd >> 5;
  for (int t = 0; t < ntiles; ++t) {
    const int cur = t & 1, nxt = cur ^ 1;
    const bool has_next = (t + 1) < ntiles;
    if (has_next) {                       // kick off prefetch of tile t+1
      issueB(nxt, (t + 1) << 5);
      loadA((t + 1) << 5);
    }

    // ---- fragments from buffer `cur` (ISA 16-bit A/B VGPR layouts) ----
    Frag a[2], b[4];
    #pragma unroll
    for (int mi = 0; mi < 2; ++mi) {
      const int r = wm * 32 + mi * 16 + lr;
      #pragma unroll
      for (int v = 0; v < 8; ++v) {
        const int k0 = ((v & 3) << 1) + ((v >> 2) << 4) + (hfl << 3);
        a[mi].d[v] = *(const unsigned int*)&sA[cur][r][k0];
      }
    }
    #pragma unroll
    for (int ni = 0; ni < 4; ++ni) {
      const int c = wn * 64 + ni * 16 + lr;
      #pragma unroll
      for (int v = 0; v < 8; ++v) {
        const int k0 = ((v & 3) << 1) + ((v >> 2) << 4) + (hfl << 3);
        b[ni].d[v] = *(const unsigned int*)&sB[cur][c][k0];
      }
    }

    // ---- 8 WMMAs per wave per K-step ----
    #pragma unroll
    for (int mi = 0; mi < 2; ++mi)
      #pragma unroll
      for (int ni = 0; ni < 4; ++ni)
        acc[mi][ni] = __builtin_amdgcn_wmma_f32_16x16x32_bf16(
            false, a[mi].v, false, b[ni].v, (short)0, acc[mi][ni], false, false);

    // ---- commit prefetched tile t+1 into buffer `nxt` ----
    if (has_next) {
#if !GEMM_USE_ASYNC
      commitB(nxt);
#endif
      storeA(nxt);
    }
#if GEMM_USE_ASYNC
    wait_async0();                        // B tile (ASYNCcnt) done
#endif
    __syncthreads();                      // A tile (DScnt) done + block sync
  }

  // ---- epilogue: bias (+relu) (+residual), row-pointer addressing ----
  #pragma unroll
  for (int mi = 0; mi < 2; ++mi) {
    const int gm0 = bm + wm * 32 + mi * 16 + (hfl << 3);
    #pragma unroll
    for (int ni = 0; ni < 4; ++ni) {
      const int gn = bn + wn * 64 + ni * 16 + lr;
      const float bv = bias[gn];
      float* cp = C + (size_t)gm0 * N + gn;
      const float* rp = resid ? resid + (size_t)gm0 * N + gn : nullptr;
      #pragma unroll
      for (int vr = 0; vr < 8; ++vr) {
        float val = acc[mi][ni][vr] + bv;
        if (relu) val = fmaxf(val, 0.0f);
        if (rp)   val += rp[(size_t)vr * N];
        cp[(size_t)vr * N] = val;
      }
    }
  }
}

// ---------------------------------------------------------------------------
// Gating: logits = latent @ w_gate [512x8]; top-2 softmax -> gates[B,8]
// ---------------------------------------------------------------------------
__global__ __launch_bounds__(256)
void gate_topk(const float* __restrict__ lat, const float* __restrict__ wg,
               float* __restrict__ gates)
{
  __shared__ float sW[LATD * NEXP];
  const int tid = threadIdx.x;
  for (int i = tid; i < LATD * NEXP; i += 256) sW[i] = wg[i];
  __syncthreads();

  const int b = blockIdx.x * 256 + tid;
  const float* lb = lat + (size_t)b * LATD;
  float logit[NEXP];
  #pragma unroll
  for (int e = 0; e < NEXP; ++e) logit[e] = 0.0f;
  for (int k = 0; k < LATD; ++k) {
    const float lv = lb[k];
    #pragma unroll
    for (int e = 0; e < NEXP; ++e) logit[e] = fmaf(lv, sW[k * NEXP + e], logit[e]);
  }
  int i1 = 0; float v1 = logit[0];
  #pragma unroll
  for (int e = 1; e < NEXP; ++e) if (logit[e] > v1) { v1 = logit[e]; i1 = e; }
  int i2 = -1; float v2 = -3.4e38f;
  #pragma unroll
  for (int e = 0; e < NEXP; ++e) if (e != i1 && logit[e] > v2) { v2 = logit[e]; i2 = e; }
  const float g2  = __expf(v2 - v1);
  const float inv = 1.0f / (1.0f + g2);
  #pragma unroll
  for (int e = 0; e < NEXP; ++e) {
    float g = 0.0f;
    if (e == i1) g = inv;
    else if (e == i2) g = g2 * inv;
    gates[(size_t)b * NEXP + e] = g;
  }
}

// y[b,o] (+)= gate[b,e] * (h[b,:] @ Wf_e[:,19] + bf_e[o])
__global__ __launch_bounds__(256)
void expert_combine(const float* __restrict__ h, const float* __restrict__ Wf,
                    const float* __restrict__ bf, const float* __restrict__ gates,
                    int e, float* __restrict__ y, int init)
{
  const int i = blockIdx.x * 256 + threadIdx.x;
  if (i >= BT * OUTD) return;
  const int b = i / OUTD, o = i - b * OUTD;
  const float* hb = h + (size_t)b * LATD;
  float s = 0.0f;
  for (int k = 0; k < LATD; ++k) s = fmaf(hb[k], Wf[(size_t)k * OUTD + o], s);
  const float val = (s + bf[o]) * gates[(size_t)b * NEXP + e];
  if (init) y[i] = val; else y[i] += val;
}

// out[b,19] = h[b,:] @ val_Wf[:,1] + val_bf
__global__ __launch_bounds__(256)
void value_final(const float* __restrict__ h, const float* __restrict__ Wf,
                 const float* __restrict__ bf, float* __restrict__ out)
{
  const int b = blockIdx.x * 256 + threadIdx.x;
  const float* hb = h + (size_t)b * OBSD;
  float s = 0.0f;
  for (int k = 0; k < OBSD; ++k) s = fmaf(hb[k], Wf[k], s);
  out[(size_t)b * (OUTD + 1) + OUTD] = s + bf[0];
}

// out[b,0:19] = (y == 0) ? EPS : y    (SparseDispatcher.combine quirk)
__global__ __launch_bounds__(256)
void finalize_y(const float* __restrict__ y, float* __restrict__ out)
{
  const int i = blockIdx.x * 256 + threadIdx.x;
  if (i >= BT * OUTD) return;
  const int b = i / OUTD, o = i - b * OUTD;
  const float v = y[i];
  out[(size_t)b * (OUTD + 1) + o] = (v == 0.0f) ? EPSF : v;
}

// ---------------------------------------------------------------------------
extern "C" void kernel_launch(void* const* d_in, const int* in_sizes, int n_in,
                              void* d_out, int out_size, void* d_ws, size_t ws_size,
                              hipStream_t stream)
{
  (void)in_sizes; (void)n_in; (void)out_size; (void)ws_size;

  const float* obs    = (const float*)d_in[0];
  const float* enc_W1 = (const float*)d_in[1];
  const float* enc_b1 = (const float*)d_in[2];
  const float* enc_W2 = (const float*)d_in[3];
  const float* enc_b2 = (const float*)d_in[4];
  const float* enc_Wf = (const float*)d_in[5];
  const float* enc_bf = (const float*)d_in[6];
  const float* exp_W1 = (const float*)d_in[7];
  const float* exp_b1 = (const float*)d_in[8];
  const float* exp_W2 = (const float*)d_in[9];
  const float* exp_b2 = (const float*)d_in[10];
  const float* exp_Wf = (const float*)d_in[11];
  const float* exp_bf = (const float*)d_in[12];
  const float* val_W1 = (const float*)d_in[13];
  const float* val_b1 = (const float*)d_in[14];
  const float* val_W2 = (const float*)d_in[15];
  const float* val_b2 = (const float*)d_in[16];
  const float* val_Wf = (const float*)d_in[17];
  const float* val_bf = (const float*)d_in[18];
  const float* w_gate = (const float*)d_in[19];
  float* out = (float*)d_out;

  // ---- carve workspace ----
  char* p = (char*)d_ws;
  auto carveF = [&](size_t n) { float* q = (float*)p; p += n * sizeof(float); return q; };
  auto carveH = [&](size_t n) { __bf16* q = (__bf16*)p; p += n * sizeof(__bf16); return q; };

  float* a0    = carveF((size_t)BT * OBSD);
  float* a1    = carveF((size_t)BT * HIDD);
  float* a2    = carveF((size_t)BT * OBSD);
  float* lat   = carveF((size_t)BT * LATD);
  float* eh0   = carveF((size_t)BT * LATD);
  float* eh1   = carveF((size_t)BT * LATD);
  float* gates = carveF((size_t)BT * NEXP);
  float* ybuf  = carveF((size_t)BT * OUTD);

  __bf16* t_encW1 = carveH((size_t)NBLK * OBSD * HIDD);
  __bf16* t_encW2 = carveH((size_t)NBLK * HIDD * OBSD);
  __bf16* t_encWf = carveH((size_t)OBSD * LATD);
  __bf16* t_expW1 = carveH((size_t)NEXP * NBLK * LATD * HIDD);
  __bf16* t_expW2 = carveH((size_t)NEXP * NBLK * HIDD * LATD);
  __bf16* t_valW1 = carveH((size_t)NBLK * OBSD * HIDD);
  __bf16* t_valW2 = carveH((size_t)NBLK * HIDD * OBSD);

  // ---- pre-transpose + bf16-convert all WMMA weights (once per launch) ----
  auto xpose = [&](const float* W, __bf16* Wt, int Kd, int N, int L) {
    dim3 g(N / 32, Kd / 32, L);
    transpose_bf16<<<g, 256, 0, stream>>>(W, Wt, Kd, N);
  };
  xpose(enc_W1, t_encW1, OBSD, HIDD, NBLK);
  xpose(enc_W2, t_encW2, HIDD, OBSD, NBLK);
  xpose(enc_Wf, t_encWf, OBSD, LATD, 1);
  xpose(exp_W1, t_expW1, LATD, HIDD, NEXP * NBLK);
  xpose(exp_W2, t_expW2, HIDD, LATD, NEXP * NBLK);
  xpose(val_W1, t_valW1, OBSD, HIDD, NBLK);
  xpose(val_W2, t_valW2, HIDD, OBSD, NBLK);

  auto gemm = [&](const float* A, const __bf16* Wt, const float* bias,
                  const float* resid, float* C, int M, int N, int Kd, int relu) {
    dim3 grid(N / 128, M / 128);
    gemm_bias_act<<<grid, 256, 0, stream>>>(A, Wt, bias, resid, C, M, N, Kd, relu);
  };

  // ---- encoder ResNet: obs -> latent ----
  const float* cur = obs;
  float* nxt = a0;
  for (int i = 0; i < NBLK; ++i) {
    gemm(cur, t_encW1 + (size_t)i * OBSD * HIDD, enc_b1 + (size_t)i * HIDD,
         nullptr, a1, BT, HIDD, OBSD, 1);
    gemm(a1,  t_encW2 + (size_t)i * HIDD * OBSD, enc_b2 + (size_t)i * OBSD,
         cur, nxt, BT, OBSD, HIDD, 0);
    cur = nxt;
    nxt = (cur == a0) ? a2 : a0;
  }
  gemm(cur, t_encWf, enc_bf, nullptr, lat, BT, LATD, OBSD, 0);

  // ---- top-2 gating ----
  gate_topk<<<BT / 256, 256, 0, stream>>>(lat, w_gate, gates);

  // ---- experts (dense, matching reference) ----
  const int cblk = (BT * OUTD + 255) / 256;
  for (int e = 0; e < NEXP; ++e) {
    const float* ec = lat;
    float* en = eh0;
    for (int i = 0; i < NBLK; ++i) {
      const size_t wi = (size_t)(e * NBLK + i);
      gemm(ec, t_expW1 + wi * LATD * HIDD, exp_b1 + wi * HIDD,
           nullptr, a1, BT, HIDD, LATD, 1);
      gemm(a1, t_expW2 + wi * HIDD * LATD, exp_b2 + wi * LATD,
           ec, en, BT, LATD, HIDD, 0);
      ec = en;
      en = (ec == eh0) ? eh1 : eh0;
    }
    expert_combine<<<cblk, 256, 0, stream>>>(
        ec, exp_Wf + (size_t)e * LATD * OUTD, exp_bf + (size_t)e * OUTD,
        gates, e, ybuf, e == 0 ? 1 : 0);
  }

  // ---- value ResNet: obs -> 1 (reuses a0/a1/a2) ----
  cur = obs;
  nxt = a0;
  for (int i = 0; i < NBLK; ++i) {
    gemm(cur, t_valW1 + (size_t)i * OBSD * HIDD, val_b1 + (size_t)i * HIDD,
         nullptr, a1, BT, HIDD, OBSD, 1);
    gemm(a1,  t_valW2 + (size_t)i * HIDD * OBSD, val_b2 + (size_t)i * OBSD,
         cur, nxt, BT, OBSD, HIDD, 0);
    cur = nxt;
    nxt = (cur == a0) ? a2 : a0;
  }
  value_final<<<BT / 256, 256, 0, stream>>>(cur, val_Wf, val_bf, out);

  // ---- y with EPS quirk -> out[:, 0:19] ----
  finalize_y<<<cblk, 256, 0, stream>>>(ybuf, out);
}
